// LeViT_SubsampleAttention_47888885350865
// MI455X (gfx1250) — matrix operand order repaired and checked by reading, not compiled, not run on hardware
//
#include <hip/hip_runtime.h>
#include <hip/hip_bf16.h>

// ---------------------------------------------------------------------------
// LeViT subsample attention for MI455X (gfx1250, wave32, WMMA).
// All heavy math runs on v_wmma_f32_16x16x32_f16 (f16 in, f32 accumulate).
// ---------------------------------------------------------------------------

typedef __attribute__((ext_vector_type(16))) _Float16 v16h;
typedef __attribute__((ext_vector_type(8)))  float    v8f;

#define DIM      64
#define HEADS    12
#define DD       768        // DIM*HEADS
#define OUT_DIM  512
#define BB       32
#define HW       784        // 28*28
#define HWP      800        // padded to 50*16
#define NQ       196        // 14*14
#define NQP      208        // padded to 13*16
#define CIN      384
#define KVC      3840       // 5*D
#define LDK      40         // LDS row stride in halves: 80B -> 16B aligned rows

static __device__ __forceinline__ v8f zero8() {
    v8f z;
#pragma unroll
    for (int i = 0; i < 8; ++i) z[i] = 0.0f;
    return z;
}

// ---------------------------------------------------------------------------
// Generic tiled WMMA GEMM:  C[z] = A[z] (MxK, row major) * B[z] (KxN, row major)
// Requirements used here: K % 32 == 0, lda == K, ldb == N, ldc == N.
// Block tile 128x128, 8 waves (4x2), each wave owns 32x64 (2x4 16x16 frags).
// Double-buffered LDS (one barrier per k-step); L2 prefetch of the k+2 tile.
// EXACT=true compiles out all M/N edge masks (M%128==0 && N%128==0).
// ---------------------------------------------------------------------------
template <typename TA, typename TB, bool CHALF, bool EXACT>
__global__ __launch_bounds__(256)
void wmma_gemm(const TA* __restrict__ A, const TB* __restrict__ B,
               void* __restrict__ Cv, int M, int N, int K,
               long strideA, long strideB, long strideC)
{
    const int tid  = threadIdx.x;
    const int lane = tid & 31;
    const int wid  = tid >> 5;
    const int lrow = lane & 15;
    const int hi   = lane >> 4;
    const int wm   = wid & 3;      // wave row   -> *32
    const int wn   = wid >> 2;     // wave col   -> *64
    const int m0   = blockIdx.y * 128;
    const int n0   = blockIdx.x * 128;
    const long zb  = blockIdx.z;

    A += zb * strideA;
    B += zb * strideB;

    __shared__ __align__(16) _Float16 As[2][128][LDK];   // [m][k]
    __shared__ __align__(16) _Float16 Bs[2][128][LDK];   // [n][k] (transposed)

    // staging thread roles
    const int  ar  = tid >> 1;          // A: 2 threads/row, 16 cols each
    const int  acs = (tid & 1) * 16;
    const bool aok = EXACT || ((m0 + ar) < M);
    const int  bk  = tid >> 3;          // B: thread t -> k=t>>3, cols (t&7)*16
    const int  bcs = (tid & 7) * 16;

    auto stage = [&](int buf, int k0) {
        const TA* ap = A + (long)(m0 + ar) * K + (k0 + acs);
#pragma unroll
        for (int j = 0; j < 16; ++j)
            As[buf][ar][acs + j] = aok ? (_Float16)ap[j] : (_Float16)0.0f;
        const TB* bp = B + (long)(k0 + bk) * N + (n0 + bcs);
#pragma unroll
        for (int j = 0; j < 16; ++j) {
            const bool ok = EXACT || ((n0 + bcs + j) < N);
            Bs[buf][bcs + j][bk] = ok ? (_Float16)bp[j] : (_Float16)0.0f;
        }
    };

    v8f acc[2][4];
#pragma unroll
    for (int mt = 0; mt < 2; ++mt)
#pragma unroll
        for (int nt = 0; nt < 4; ++nt) acc[mt][nt] = zero8();

    const int nk = K / 32;
    stage(0, 0);
    __syncthreads();

    for (int kt = 0; kt < nk; ++kt) {
        const int cur = kt & 1;
        if (kt + 1 < nk) {
            if (kt + 2 < nk) {
                // gfx1250 global_prefetch_b8: pull the k+2 tile toward L2
                if (aok)
                    __builtin_prefetch(A + (long)(m0 + ar) * K + ((kt + 2) * 32 + acs), 0, 3);
                __builtin_prefetch(B + (long)((kt + 2) * 32 + bk) * N + (n0 + bcs), 0, 3);
            }
            stage(cur ^ 1, (kt + 1) * 32);   // write other buffer: no conflict
        }

        // ---- fragments (contiguous k = 16*hi + j, same in A and B) from buffer `cur`
        v16h afrag[2], bfrag[4];
#pragma unroll
        for (int mt = 0; mt < 2; ++mt) {
            const _Float16* ap = &As[cur][wm * 32 + mt * 16 + lrow][0];
#pragma unroll
            for (int j = 0; j < 16; ++j) afrag[mt][j] = ap[j + 16 * hi];
        }
#pragma unroll
        for (int nt = 0; nt < 4; ++nt) {
            const _Float16* bp = &Bs[cur][wn * 64 + nt * 16 + lrow][0];
#pragma unroll
            for (int j = 0; j < 16; ++j) bfrag[nt][j] = bp[j + 16 * hi];
        }
#pragma unroll
        for (int mt = 0; mt < 2; ++mt)
#pragma unroll
            for (int nt = 0; nt < 4; ++nt)
                acc[mt][nt] = __builtin_amdgcn_wmma_f32_16x16x32_f16(
                    false, afrag[mt], false, bfrag[nt],
                    (short)0, acc[mt][nt], false, false);
        __syncthreads();
    }

    // ---- epilogue: D layout m = r + 8*hi, n = lane%16
    float*    Cf = (float*)Cv;
    _Float16* Ch = (_Float16*)Cv;
#pragma unroll
    for (int mt = 0; mt < 2; ++mt)
#pragma unroll
        for (int nt = 0; nt < 4; ++nt)
#pragma unroll
            for (int r = 0; r < 8; ++r) {
                const int row = m0 + wm * 32 + mt * 16 + r + 8 * hi;
                const int col = n0 + wn * 64 + nt * 16 + lrow;
                if (EXACT || (row < M && col < N)) {
                    const long off = zb * strideC + (long)row * N + col;
                    if (CHALF) Ch[off] = (_Float16)acc[mt][nt][r];
                    else       Cf[off] = acc[mt][nt][r];
                }
            }
}

// ---------------------------------------------------------------------------
// Gather strided q input:  Qx[b*196+i][c] = x[b][2*(i/14)][2*(i%14)][c]
// ---------------------------------------------------------------------------
__global__ void gather_q(const float* __restrict__ x, float* __restrict__ Qx, int total)
{
    int idx = blockIdx.x * blockDim.x + threadIdx.x;
    if (idx >= total) return;
    int c = idx % CIN;
    int r = idx / CIN;
    int b = r / NQ, i = r % NQ;
    int hh = (i / 14) * 2, ww = (i % 14) * 2;
    Qx[idx] = x[(((long)b * 28 + hh) * 28 + ww) * CIN + c];
}

// ---------------------------------------------------------------------------
// BatchNorm column statistics over an f16 matrix (row major R x Cn)
// ---------------------------------------------------------------------------
__global__ void col_stats(const _Float16* __restrict__ X, float* __restrict__ sums,
                          float* __restrict__ sumsq, int R, int Cn, int chunk)
{
    int col = blockIdx.x * blockDim.x + threadIdx.x;
    if (col >= Cn) return;
    int r0 = blockIdx.y * chunk;
    int r1 = r0 + chunk; if (r1 > R) r1 = R;
    float s = 0.0f, s2 = 0.0f;
    for (int r = r0; r < r1; ++r) {
        float v = (float)X[(long)r * Cn + col];
        s += v; s2 += v * v;
    }
    atomicAdd(&sums[col], s);
    atomicAdd(&sumsq[col], s2);
}

__global__ void finalize_stats(const float* __restrict__ sums, const float* __restrict__ sumsq,
                               const float* __restrict__ scale, float* __restrict__ mean,
                               float* __restrict__ inv, int Cn, float Rf)
{
    int c = blockIdx.x * blockDim.x + threadIdx.x;
    if (c >= Cn) return;
    float m   = sums[c] / Rf;
    float var = sumsq[c] / Rf - m * m;
    mean[c] = m;
    inv[c]  = rsqrtf(var + 1e-5f) * scale[c];
}

// ---------------------------------------------------------------------------
// Normalize kv and repack:  Kh[b,h,p(0..799),d64]  Vh[b,h,p(0..799),e256]
// (pad rows 784..799 left untouched: masked in attention / multiplied by 0)
// ---------------------------------------------------------------------------
__global__ void pack_kv(const _Float16* __restrict__ kv, const float* __restrict__ mean,
                        const float* __restrict__ inv, _Float16* __restrict__ Kh,
                        _Float16* __restrict__ Vh, long total)
{
    long idx = (long)blockIdx.x * blockDim.x + threadIdx.x;
    if (idx >= total) return;
    int  c = (int)(idx % KVC);
    long r = idx / KVC;
    int  b = (int)(r / HW), p = (int)(r % HW);
    float v = ((float)kv[idx] - mean[c]) * inv[c];
    if (c < DD) {
        int h = c >> 6, d = c & 63;
        Kh[(((long)b * HEADS + h) * HWP + p) * DIM + d] = (_Float16)v;
    } else {
        int cc = c - DD;
        int h = cc >> 8, e = cc & 255;
        Vh[(((long)b * HEADS + h) * HWP + p) * 256 + e] = (_Float16)v;
    }
}

__global__ void pack_q(const _Float16* __restrict__ q, const float* __restrict__ mean,
                       const float* __restrict__ inv, _Float16* __restrict__ Qh, int total)
{
    int idx = blockIdx.x * blockDim.x + threadIdx.x;
    if (idx >= total) return;
    int c = idx % DD;
    int r = idx / DD;
    int b = r / NQ, i = r % NQ;
    int h = c >> 6, d = c & 63;
    float v = ((float)q[idx] - mean[c]) * inv[c];
    Qh[(((long)b * HEADS + h) * NQ + i) * DIM + d] = (_Float16)v;
}

// ---------------------------------------------------------------------------
// Scores + softmax. One block per (b*h, 16-row q tile).
//   S(16 x 800) = Q(16x64) * K^T   via 2 WMMA ksteps, frags loaded straight
//   from global with contiguous k = 16*hi + j per lane, then row softmax -> P (f16).
// Padding is handled STRUCTURALLY: the k-position pad tile (cols 784..799,
// i.e. tile ct==49) is never computed -- it is initialized to -3e38 with one
// store per thread.  The main loop (ct in [wid,49)) is therefore fully valid:
// unconditional bias loads, zero exec-mask divergence around the WMMA.
// ---------------------------------------------------------------------------
__global__ __launch_bounds__(256)
void attn_softmax(const _Float16* __restrict__ Qh, const _Float16* __restrict__ Kh,
                  const float* __restrict__ bias, _Float16* __restrict__ P)
{
    const int bh   = blockIdx.y;   // 0..383
    const int qt   = blockIdx.x;   // 0..12
    const int tid  = threadIdx.x;
    const int lane = tid & 31;
    const int wid  = tid >> 5;
    const int lrow = lane & 15;
    const int hi   = lane >> 4;

    __shared__ float S[16][HWP];   // 51200 B

    // pad columns 784..799: 16x16 block, exactly one element per thread
    S[tid >> 4][HW + (tid & 15)] = -3.0e38f;

    // A fragments: q rows qt*16..+15 (clamped), K-dim split in 2 steps of 32
    int qrow = qt * 16 + lrow; if (qrow > NQ - 1) qrow = NQ - 1;
    const _Float16* qp = Qh + ((long)bh * NQ + qrow) * DIM;
    v16h a[2];
#pragma unroll
    for (int s = 0; s < 2; ++s) {
#pragma unroll
        for (int j = 0; j < 16; ++j) a[s][j] = qp[s * 32 + j + 16 * hi];
    }

    const _Float16* kbase = Kh + (long)bh * HWP * DIM;
    for (int ct = wid; ct < HW / 16; ct += 8) {   // tiles 0..48: all valid
        v8f accv = zero8();
        const int p = ct * 16 + lrow;             // B frag: n = lane%16
#pragma unroll
        for (int s = 0; s < 2; ++s) {
            v16h b;
            const _Float16* kp = kbase + (long)p * DIM + s * 32 + 16 * hi;
#pragma unroll
            for (int j = 0; j < 16; ++j) b[j] = kp[j];
            accv = __builtin_amdgcn_wmma_f32_16x16x32_f16(
                false, a[s], false, b, (short)0, accv, false, false);
        }
        const int col = ct * 16 + lrow;           // always < 784
#pragma unroll
        for (int r = 0; r < 8; ++r) {
            const int m  = r + 8 * hi;
            int qr = qt * 16 + m; if (qr > NQ - 1) qr = NQ - 1;
            S[m][col] = accv[r] * 0.125f + bias[(long)qr * HW + col];
        }
    }
    __syncthreads();

    // softmax: wave w handles rows 2w, 2w+1 (wave32 shuffle reductions)
    for (int rr = 0; rr < 2; ++rr) {
        const int row = wid * 2 + rr;
        float mx = -3.0e38f;
        for (int c = lane; c < HWP; c += 32) mx = fmaxf(mx, S[row][c]);
#pragma unroll
        for (int o = 16; o > 0; o >>= 1) mx = fmaxf(mx, __shfl_xor(mx, o, 32));
        float sum = 0.0f;
        for (int c = lane; c < HWP; c += 32) {
            float e = __expf(S[row][c] - mx);
            S[row][c] = e;
            sum += e;
        }
#pragma unroll
        for (int o = 16; o > 0; o >>= 1) sum += __shfl_xor(sum, o, 32);
        const float invs = 1.0f / sum;
        _Float16* pp = P + ((long)bh * NQP + qt * 16 + row) * HWP;
        for (int c = lane; c < HWP; c += 32) pp[c] = (_Float16)(S[row][c] * invs);
    }
}

// ---------------------------------------------------------------------------
// hard_swish + layout gather: O[b,h,i,e] (f32) -> o_final[b*196+i][h*256+e] f16
// ---------------------------------------------------------------------------
__global__ void hswish_gather(const float* __restrict__ O, _Float16* __restrict__ out, int total)
{
    int idx = blockIdx.x * blockDim.x + threadIdx.x;
    if (idx >= total) return;
    int c = idx % (4 * DD);
    int r = idx / (4 * DD);
    int b = r / NQ, i = r % NQ;
    int h = c >> 8, e = c & 255;
    float v  = O[(((long)(b * HEADS + h)) * NQP + i) * 256 + e];
    float hs = v * fminf(fmaxf(v + 3.0f, 0.0f), 6.0f) * (1.0f / 6.0f);
    out[idx] = (_Float16)hs;
}

// ---------------------------------------------------------------------------
// Host-side orchestration (graph-capture safe: only stream ops)
// ---------------------------------------------------------------------------
extern "C" void kernel_launch(void* const* d_in, const int* in_sizes, int n_in,
                              void* d_out, int out_size, void* d_ws, size_t ws_size,
                              hipStream_t stream)
{
    (void)in_sizes; (void)n_in; (void)out_size; (void)ws_size;
    const float* x         = (const float*)d_in[0];   // 32*28*28*384
    const float* W_qkv     = (const float*)d_in[1];   // 384*3840
    const float* bn1_scale = (const float*)d_in[2];   // 3840
    const float* W_q       = (const float*)d_in[3];   // 384*768
    const float* bn2_scale = (const float*)d_in[4];   // 768
    const float* attn_bias = (const float*)d_in[5];   // 196*784
    const float* W_out     = (const float*)d_in[6];   // 3072*512
    float* out = (float*)d_out;                       // 6272*512

    // ---- workspace layout (~539 MB; P aliases dead kv_raw region) ----
    char*  base = (char*)d_ws;
    size_t off  = 0;
    auto alloc = [&](size_t bytes) -> void* {
        void* p = base + off;
        off = (off + bytes + 255) & ~(size_t)255;
        return p;
    };
    const int MKV = BB * HW;   // 25088
    const int MQ  = BB * NQ;   // 6272
    _Float16* kv_raw  = (_Float16*)alloc((size_t)MKV * KVC * 2);       // 192.7 MB
    _Float16* q_raw   = (_Float16*)alloc((size_t)MQ * DD * 2);         //   9.6 MB
    float*    Qx      = (float*)   alloc((size_t)MQ * CIN * 4);        //   9.6 MB
    float*    stats   = (float*)   alloc((size_t)(4 * KVC + 4 * DD) * 4);
    _Float16* Kh      = (_Float16*)alloc((size_t)BB * HEADS * HWP * DIM * 2);  // 39.3 MB
    _Float16* Vh      = (_Float16*)alloc((size_t)BB * HEADS * HWP * 256 * 2);  //157.3 MB
    _Float16* Qh      = (_Float16*)alloc((size_t)BB * HEADS * NQ * DIM * 2);   //  9.6 MB
    float*    O       = (float*)   alloc((size_t)BB * HEADS * NQP * 256 * 4);  // 81.8 MB
    _Float16* o_final = (_Float16*)alloc((size_t)MQ * 4 * DD * 2);             // 38.5 MB
    _Float16* P       = kv_raw;  // alias: kv_raw is dead once pack_kv completes

    float* kv_sum   = stats;
    float* kv_sumsq = stats + KVC;
    float* kv_mean  = stats + 2 * KVC;
    float* kv_inv   = stats + 3 * KVC;
    float* q_sum    = stats + 4 * KVC;
    float* q_sumsq  = q_sum + DD;
    float* q_mean   = q_sum + 2 * DD;
    float* q_inv    = q_sum + 3 * DD;

    hipMemsetAsync(stats, 0, (size_t)(4 * KVC + 4 * DD) * 4, stream);

    // q input gather
    {
        int total = MQ * CIN;
        gather_q<<<(total + 255) / 256, 256, 0, stream>>>(x, Qx, total);
    }
    // GEMM1: kv_raw = x @ W_qkv   (25088 x 384 x 3840), f16 output, exact tiles
    wmma_gemm<float, float, true, true><<<dim3(KVC / 128, MKV / 128, 1), 256, 0, stream>>>(
        x, W_qkv, kv_raw, MKV, KVC, CIN, 0, 0, 0);
    // GEMM2: q_raw = Qx @ W_q     (6272 x 384 x 768), f16 output, exact tiles
    wmma_gemm<float, float, true, true><<<dim3(DD / 128, MQ / 128, 1), 256, 0, stream>>>(
        Qx, W_q, q_raw, MQ, DD, CIN, 0, 0, 0);

    // BatchNorm statistics
    col_stats<<<dim3((KVC + 255) / 256, (MKV + 255) / 256), 256, 0, stream>>>(
        kv_raw, kv_sum, kv_sumsq, MKV, KVC, 256);
    col_stats<<<dim3((DD + 255) / 256, (MQ + 255) / 256), 256, 0, stream>>>(
        q_raw, q_sum, q_sumsq, MQ, DD, 256);
    finalize_stats<<<(KVC + 255) / 256, 256, 0, stream>>>(
        kv_sum, kv_sumsq, bn1_scale, kv_mean, kv_inv, KVC, (float)MKV);
    finalize_stats<<<(DD + 255) / 256, 256, 0, stream>>>(
        q_sum, q_sumsq, bn2_scale, q_mean, q_inv, DD, (float)MQ);

    // Normalize + repack
    {
        long total = (long)MKV * KVC;
        pack_kv<<<(int)((total + 255) / 256), 256, 0, stream>>>(
            kv_raw, kv_mean, kv_inv, Kh, Vh, total);
    }
    {
        int total = MQ * DD;
        pack_q<<<(total + 255) / 256, 256, 0, stream>>>(q_raw, q_mean, q_inv, Qh, total);
    }

    // Scores + softmax -> P  (P aliases kv_raw, which is now dead)
    attn_softmax<<<dim3(NQP / 16, BB * HEADS), 256, 0, stream>>>(Qh, Kh, attn_bias, P);

    // O = P @ V, batched over 384 (b,h):  208 x 800 x 256 (M not tile-exact)
    wmma_gemm<_Float16, _Float16, false, false><<<dim3(2, 2, BB * HEADS), 256, 0, stream>>>(
        P, Vh, O, NQP, 256, HWP,
        (long)NQP * HWP, (long)HWP * 256, (long)NQP * 256);

    // hard_swish + gather to (6272 x 3072) f16
    {
        int total = MQ * 4 * DD;
        hswish_gather<<<(total + 255) / 256, 256, 0, stream>>>(O, o_final, total);
    }

    // Output GEMM: (6272 x 3072) @ (3072 x 512) -> d_out f32, exact tiles
    wmma_gemm<_Float16, float, false, true><<<dim3(OUT_DIM / 128, MQ / 128, 1), 256, 0, stream>>>(
        o_final, W_out, out, MQ, OUT_DIM, 4 * DD, 0, 0, 0);
}
